// MOE_10840497455302
// MI455X (gfx1250) — matrix operand order repaired and checked by reading, not compiled, not run on hardware
//
#include <hip/hip_runtime.h>
#include <hip/hip_bf16.h>
#include <math.h>

#define B_ROWS 8192
#define DV_DIM 768
#define G_DIM  2048
#define D_DIM  1024
#define E_NUM  8
#define BM     16
#define H1LD   (D_DIM + 8)

typedef __attribute__((ext_vector_type(16))) __bf16 v16bf;
typedef __attribute__((ext_vector_type(8)))  float  v8f;

union ABFrag { v16bf v; uint4 u[2]; };
union BF4    { __bf16 h[4]; uint2 u; };

__device__ __forceinline__ v8f wmma_bf16(v16bf a, v16bf b, v8f c) {
  return __builtin_amdgcn_wmma_f32_16x16x32_bf16(false, a, false, b, (short)0, c,
                                                 false, false);
}

// A fragment: 16x32 bf16 tile from row-major source, p = &A[row0][k0], stride ld.
// ISA layout: lanes 0-15 hold K 0-7 & 16-23, lanes 16-31 hold K 8-15 & 24-31.
__device__ __forceinline__ v16bf frag_a(const __bf16* p, int ld, int lane) {
  const __bf16* q = p + (size_t)(lane & 15) * ld + ((lane >> 4) << 3);
  ABFrag f;
  f.u[0] = *(const uint4*)(q);
  f.u[1] = *(const uint4*)(q + 16);
  return f.v;
}

// B fragment: 32x16 bf16 tile taken from TRANSPOSED weights Wt[n][k] (stride ld),
// p = &Wt[n0][k0]. Lanes 0-15: K 0-7 & 16-23 of col n; lanes 16-31: K 8-15 & 24-31.
__device__ __forceinline__ v16bf frag_b(const __bf16* p, int ld, int lane) {
  const __bf16* q = p + (size_t)(lane & 15) * ld + ((lane >> 4) << 3);
  ABFrag f;
  f.u[0] = *(const uint4*)(q);
  f.u[1] = *(const uint4*)(q + 16);
  return f.v;
}

// ---------------- conversion kernels ----------------

__global__ void cvt_kernel(const float* __restrict__ src, __bf16* __restrict__ dst,
                           int n4) {  // n4 = n/4
  int i = blockIdx.x * blockDim.x + threadIdx.x;
  int stride = gridDim.x * blockDim.x;
  for (; i < n4; i += stride) {
    float4 f = ((const float4*)src)[i];
    BF4 o;
    o.h[0] = (__bf16)f.x; o.h[1] = (__bf16)f.y;
    o.h[2] = (__bf16)f.z; o.h[3] = (__bf16)f.w;
    ((uint2*)dst)[i] = o.u;
  }
}

// Transpose [R][C] f32 -> [C][R] bf16 (batched over blockIdx.z). R,C multiples of 32.
__global__ __launch_bounds__(256) void tcvt_kernel(const float* __restrict__ src,
                                                   __bf16* __restrict__ dst,
                                                   int R, int C) {
  __shared__ float tile[32][33];
  size_t zo = (size_t)blockIdx.z * R * C;
  src += zo; dst += zo;
  int c0 = blockIdx.x * 32, r0 = blockIdx.y * 32;
  int tx = threadIdx.x & 31, ty = threadIdx.x >> 5;
  for (int i = ty; i < 32; i += 8)
    tile[i][tx] = src[(size_t)(r0 + i) * C + (c0 + tx)];
  __syncthreads();
  for (int i = ty; i < 32; i += 8)
    dst[(size_t)(c0 + i) * R + (r0 + tx)] = (__bf16)tile[tx][i];
}

// ---------------- projection: combined[b, 0:1024]=vis proj, [1024:2048]=txt proj ----

__global__ __launch_bounds__(512) void proj_kernel(
    const __bf16* __restrict__ visb, const __bf16* __restrict__ txtb,   // [B][768]
    const __bf16* __restrict__ viswt, const __bf16* __restrict__ txtwt, // [1024][768]
    const float* __restrict__ visbias, const float* __restrict__ txtbias,
    __bf16* __restrict__ combined)                                      // [B][2048]
{
  const int tid = threadIdx.x, lane = tid & 31, wave = tid >> 5;
  const int row0 = blockIdx.x * BM;
  const int half = blockIdx.y;
  const __bf16* A  = half ? txtb : visb;
  const __bf16* Wt = half ? txtwt : viswt;
  const float* bias = half ? txtbias : visbias;
  const int nb = wave * 4;
  const int l15 = lane & 15, hh = lane >> 4;

  v8f acc[4];
  v8f vzero = {0.f,0.f,0.f,0.f,0.f,0.f,0.f,0.f};
#pragma unroll
  for (int t = 0; t < 4; ++t) acc[t] = vzero;

  for (int k0 = 0; k0 < DV_DIM; k0 += 32) {
    v16bf a = frag_a(A + (size_t)row0 * DV_DIM + k0, DV_DIM, lane);
#pragma unroll
    for (int t = 0; t < 4; ++t) {
      v16bf b = frag_b(Wt + (size_t)(nb + t) * 16 * DV_DIM + k0, DV_DIM, lane);
      acc[t] = wmma_bf16(a, b, acc[t]);
    }
  }
#pragma unroll
  for (int t = 0; t < 4; ++t) {
    int col = (nb + t) * 16 + l15;
    float bb = bias[col];
#pragma unroll
    for (int v = 0; v < 8; ++v) {
      int row = v + (hh << 3);
      combined[(size_t)(row0 + row) * G_DIM + half * D_DIM + col] =
          (__bf16)(acc[t][v] + bb);
    }
  }
}

// ---------------- gate: softmax(combined @ gate_w + gate_b), one wave per row ----

__global__ __launch_bounds__(256) void gate_kernel(
    const __bf16* __restrict__ combined, const float* __restrict__ gw,
    const float* __restrict__ gb, float* __restrict__ gateo) {
  const int lane = threadIdx.x & 31, wave = threadIdx.x >> 5;
  const int b = blockIdx.x * 8 + wave;
  const __bf16* row = combined + (size_t)b * G_DIM;
  float acc[E_NUM];
#pragma unroll
  for (int e = 0; e < E_NUM; ++e) acc[e] = 0.f;
  for (int k = lane; k < G_DIM; k += 32) {
    float c = (float)row[k];
    const float* g = gw + (size_t)k * E_NUM;
#pragma unroll
    for (int e = 0; e < E_NUM; ++e) acc[e] += c * g[e];
  }
#pragma unroll
  for (int e = 0; e < E_NUM; ++e)
#pragma unroll
    for (int m = 16; m >= 1; m >>= 1) acc[e] += __shfl_xor(acc[e], m, 32);
  if (lane == 0) {
    float mx = -3.0e38f;
#pragma unroll
    for (int e = 0; e < E_NUM; ++e) { acc[e] += gb[e]; mx = fmaxf(mx, acc[e]); }
    float sum = 0.f;
#pragma unroll
    for (int e = 0; e < E_NUM; ++e) { acc[e] = __expf(acc[e] - mx); sum += acc[e]; }
    float inv = 1.0f / sum;
#pragma unroll
    for (int e = 0; e < E_NUM; ++e) gateo[(size_t)b * E_NUM + e] = acc[e] * inv;
  }
}

// ---------------- fused experts: per 16-row block, loop e: GEMM1+GELU -> LDS,
// ---------------- GEMM2 (regs), LN via shfl+ds_add_f32, gated accumulate -------

__global__ __launch_bounds__(512) void moe_expert_kernel(
    const __bf16* __restrict__ combined, // [B][2048]
    const __bf16* __restrict__ w1t,      // [E][1024][2048]
    const float*  __restrict__ b1,       // [E][1024]
    const __bf16* __restrict__ w2t,      // [E][1024][1024]
    const float*  __restrict__ b2,
    const float*  __restrict__ ln_g,
    const float*  __restrict__ ln_b,
    const float*  __restrict__ gate,     // [B][8]
    float* __restrict__ out)             // [B][1024]
{
  __shared__ __align__(16) __bf16 H1[BM][H1LD];
  __shared__ float red[BM][2];
  __shared__ float sgate[BM][E_NUM];

  const int tid = threadIdx.x;
  const int lane = tid & 31;
  const int wave = tid >> 5;           // 0..15
  const int row0 = blockIdx.x * BM;
  const int nb = wave * 4;             // 4 N-tiles of 16 cols
  const int l15 = lane & 15, hh = lane >> 4;

  if (tid < BM * E_NUM) {
    int r = tid >> 3, e = tid & 7;
    sgate[r][e] = gate[(size_t)(row0 + r) * E_NUM + e];
  }

  v8f vzero = {0.f,0.f,0.f,0.f,0.f,0.f,0.f,0.f};
  float outacc[4][8];
#pragma unroll
  for (int t = 0; t < 4; ++t)
#pragma unroll
    for (int v = 0; v < 8; ++v) outacc[t][v] = 0.f;

  for (int e = 0; e < E_NUM; ++e) {
    __syncthreads();                   // prev iter done reading H1/red
    if (tid < BM) { red[tid][0] = 0.f; red[tid][1] = 0.f; }

    // ---- Stage 1: H1 = gelu(combined @ w1[e] + b1[e]) ----
    const __bf16* w1e = w1t + (size_t)e * D_DIM * G_DIM;
    v8f acc[4];
#pragma unroll
    for (int t = 0; t < 4; ++t) acc[t] = vzero;
    for (int k0 = 0; k0 < G_DIM; k0 += 32) {
      v16bf a = frag_a(combined + (size_t)row0 * G_DIM + k0, G_DIM, lane);
#pragma unroll
      for (int t = 0; t < 4; ++t) {
        v16bf b = frag_b(w1e + (size_t)(nb + t) * 16 * G_DIM + k0, G_DIM, lane);
        acc[t] = wmma_bf16(a, b, acc[t]);
      }
    }
#pragma unroll
    for (int t = 0; t < 4; ++t) {
      int col = (nb + t) * 16 + l15;
      float bias = b1[e * D_DIM + col];
#pragma unroll
      for (int v = 0; v < 8; ++v) {
        int row = v + (hh << 3);
        float x = acc[t][v] + bias;
        float g = 0.5f * x * (1.0f + erff(x * 0.70710678118654752f));
        H1[row][col] = (__bf16)g;
      }
    }
    __syncthreads();

    // ---- Stage 2: h2 = H1 @ w2[e] + b2[e], kept in registers ----
    const __bf16* w2e = w2t + (size_t)e * D_DIM * D_DIM;
#pragma unroll
    for (int t = 0; t < 4; ++t) acc[t] = vzero;
    for (int k0 = 0; k0 < D_DIM; k0 += 32) {
      const __bf16* q = &H1[l15][k0 + (hh << 3)];
      ABFrag fa;
      fa.u[0] = *(const uint4*)(q);
      fa.u[1] = *(const uint4*)(q + 16);
#pragma unroll
      for (int t = 0; t < 4; ++t) {
        v16bf b = frag_b(w2e + (size_t)(nb + t) * 16 * D_DIM + k0, D_DIM, lane);
        acc[t] = wmma_bf16(fa.v, b, acc[t]);
      }
    }

    // ---- LayerNorm partials ----
    float vals[4][8], s[8], q2[8];
#pragma unroll
    for (int v = 0; v < 8; ++v) { s[v] = 0.f; q2[v] = 0.f; }
#pragma unroll
    for (int t = 0; t < 4; ++t) {
      int col = (nb + t) * 16 + l15;
      float bias = b2[e * D_DIM + col];
#pragma unroll
      for (int v = 0; v < 8; ++v) {
        float x = acc[t][v] + bias;
        vals[t][v] = x;
        s[v] += x; q2[v] += x * x;
      }
    }
#pragma unroll
    for (int v = 0; v < 8; ++v) {
#pragma unroll
      for (int m = 8; m >= 1; m >>= 1) {    // reduce across the 16-lane half
        s[v]  += __shfl_xor(s[v],  m, 32);
        q2[v] += __shfl_xor(q2[v], m, 32);
      }
    }
    if (l15 == 0) {
#pragma unroll
      for (int v = 0; v < 8; ++v) {
        int row = v + (hh << 3);
        atomicAdd(&red[row][0], s[v]);
        atomicAdd(&red[row][1], q2[v]);
      }
    }
    __syncthreads();

    // ---- LN + gate-weighted accumulate ----
#pragma unroll
    for (int v = 0; v < 8; ++v) {
      int row = v + (hh << 3);
      float mu  = red[row][0] * (1.0f / D_DIM);
      float var = red[row][1] * (1.0f / D_DIM) - mu * mu;
      float rs  = rsqrtf(var + 1e-5f);
      float p   = sgate[row][e];
#pragma unroll
      for (int t = 0; t < 4; ++t) {
        int col = (nb + t) * 16 + l15;
        float y = (vals[t][v] - mu) * rs * ln_g[e * D_DIM + col] + ln_b[e * D_DIM + col];
        outacc[t][v] += y * p;
      }
    }
  }

#pragma unroll
  for (int t = 0; t < 4; ++t) {
    int col = (nb + t) * 16 + l15;
#pragma unroll
    for (int v = 0; v < 8; ++v) {
      int row = v + (hh << 3);
      out[(size_t)(row0 + row) * D_DIM + col] = outacc[t][v];
    }
  }
}

// ---------------- host ----------------

extern "C" void kernel_launch(void* const* d_in, const int* in_sizes, int n_in,
                              void* d_out, int out_size, void* d_ws, size_t ws_size,
                              hipStream_t stream) {
  const float* visual = (const float*)d_in[0];
  const float* text   = (const float*)d_in[1];
  const float* vis_w  = (const float*)d_in[2];
  const float* vis_b  = (const float*)d_in[3];
  const float* txt_w  = (const float*)d_in[4];
  const float* txt_b  = (const float*)d_in[5];
  const float* gate_w = (const float*)d_in[6];
  const float* gate_b = (const float*)d_in[7];
  const float* w1     = (const float*)d_in[8];
  const float* b1     = (const float*)d_in[9];
  const float* w2     = (const float*)d_in[10];
  const float* b2     = (const float*)d_in[11];
  const float* ln_g   = (const float*)d_in[12];
  const float* ln_b   = (const float*)d_in[13];
  float* out = (float*)d_out;

  char* ws = (char*)d_ws;
  size_t off = 0;
  auto wsalloc = [&](size_t bytes) -> void* {
    void* p = ws + off;
    off = (off + bytes + 255) & ~(size_t)255;
    return p;
  };
  __bf16* visb   = (__bf16*)wsalloc((size_t)B_ROWS * DV_DIM * 2);
  __bf16* txtb   = (__bf16*)wsalloc((size_t)B_ROWS * DV_DIM * 2);
  __bf16* viswt  = (__bf16*)wsalloc((size_t)D_DIM * DV_DIM * 2);
  __bf16* txtwt  = (__bf16*)wsalloc((size_t)D_DIM * DV_DIM * 2);
  __bf16* w1t    = (__bf16*)wsalloc((size_t)E_NUM * D_DIM * G_DIM * 2);
  __bf16* w2t    = (__bf16*)wsalloc((size_t)E_NUM * D_DIM * D_DIM * 2);
  __bf16* combined = (__bf16*)wsalloc((size_t)B_ROWS * G_DIM * 2);
  float*  gatep  = (float*)wsalloc((size_t)B_ROWS * E_NUM * 4);

  cvt_kernel<<<2048, 256, 0, stream>>>(visual, visb, B_ROWS * DV_DIM / 4);
  cvt_kernel<<<2048, 256, 0, stream>>>(text,   txtb, B_ROWS * DV_DIM / 4);
  tcvt_kernel<<<dim3(D_DIM / 32, DV_DIM / 32, 1), 256, 0, stream>>>(vis_w, viswt, DV_DIM, D_DIM);
  tcvt_kernel<<<dim3(D_DIM / 32, DV_DIM / 32, 1), 256, 0, stream>>>(txt_w, txtwt, DV_DIM, D_DIM);
  tcvt_kernel<<<dim3(D_DIM / 32, G_DIM / 32, E_NUM), 256, 0, stream>>>(w1, w1t, G_DIM, D_DIM);
  tcvt_kernel<<<dim3(D_DIM / 32, D_DIM / 32, E_NUM), 256, 0, stream>>>(w2, w2t, D_DIM, D_DIM);
  proj_kernel<<<dim3(B_ROWS / BM, 2), 512, 0, stream>>>(visb, txtb, viswt, txtwt,
                                                        vis_b, txt_b, combined);
  gate_kernel<<<B_ROWS / 8, 256, 0, stream>>>(combined, gate_w, gate_b, gatep);
  moe_expert_kernel<<<B_ROWS / BM, 512, 0, stream>>>(combined, w1t, b1, w2t, b2,
                                                     ln_g, ln_b, gatep, out);
}
